// Attention_77257871721030
// MI455X (gfx1250) — matrix operand order repaired and checked by reading, not compiled, not run on hardware
//
#include <hip/hip_runtime.h>

typedef __bf16 bf16;
typedef bf16 v16bf __attribute__((ext_vector_type(16)));
typedef bf16 v8bf  __attribute__((ext_vector_type(8)));
typedef float v8f  __attribute__((ext_vector_type(8)));

constexpr int NKc = 8192;   // keys
constexpr int NQc = 8192;   // queries
constexpr int DD  = 64;     // head dim

// ---------------- preprocessing: f32 -> bf16 (scale folded into Q), V transposed ----------------

__global__ void cvt_q(const float* __restrict__ Q, bf16* __restrict__ Qb) {
  int i = blockIdx.x * 256 + threadIdx.x;
  Qb[i] = (bf16)(Q[i] * 0.125f);           // 1/sqrt(64)
}

__global__ void cvt_k(const float* __restrict__ K, bf16* __restrict__ Kb) {
  int i = blockIdx.x * 256 + threadIdx.x;
  Kb[i] = (bf16)K[i];
}

__global__ void cvt_vt(const float* __restrict__ V, bf16* __restrict__ VT) {
  int i = blockIdx.x * 256 + threadIdx.x;  // i = d*NK + k
  int d = i >> 13;                         // / 8192
  int k = i & (NKc - 1);
  VT[i] = (bf16)V[k * DD + d];
}

// ---------------- flash attention main kernel ----------------
// wave32: each wave owns 16 query rows, loops over keys in tiles of 32,
// K B-tiles double-buffered in registers (loads issued one tile ahead).

__device__ __forceinline__ void load_ktile(const bf16* __restrict__ Kb, int kb,
                                           int l15, bool hi, v16bf (&bk)[4]) {
  // B operand (32x16), k order used here: lane = column (key), 16 contiguous d-values per lane.
  const bf16* kp0 = Kb + (size_t)(kb + l15) * DD + (hi ? 16 : 0);
  const bf16* kp1 = Kb + (size_t)(kb + 16 + l15) * DD + (hi ? 16 : 0);
  bk[0] = *(const v16bf*)(kp0);            // h=0, d 0..31 slice
  bk[1] = *(const v16bf*)(kp0 + 32);       // h=0, d 32..63 slice
  bk[2] = *(const v16bf*)(kp1);            // h=1, d 0..31 slice
  bk[3] = *(const v16bf*)(kp1 + 32);       // h=1, d 32..63 slice
}

__launch_bounds__(256)
__global__ void fattn(const bf16* __restrict__ Qb, const bf16* __restrict__ Kb,
                      const bf16* __restrict__ VT, const float* __restrict__ M,
                      float* __restrict__ out) {
  __shared__ bf16 lds[8][16][32];          // per-wave 16x32 bf16 P staging (8 KB total)

  const int lane = threadIdx.x & 31;
  const int wave = threadIdx.x >> 5;
  const bool hi  = lane >= 16;
  const int l15  = lane & 15;
  const int q0   = (blockIdx.x * 8 + wave) * 16;

  // ---- load Q A-tiles (16x64 split into two 16x32 A operands) ----
  // A k order used here (same as B): lanes 0-15 row=lane, K=0-15 contiguous;
  //           lanes 16-31 same rows, K=16-31 contiguous.
  v16bf qa0, qa1;
  {
    const bf16* qp = Qb + (size_t)(q0 + l15) * DD + (hi ? 16 : 0);
    v8bf t0 = *(const v8bf*)(qp);
    v8bf t1 = *(const v8bf*)(qp + 8);
    v8bf t2 = *(const v8bf*)(qp + 32);
    v8bf t3 = *(const v8bf*)(qp + 40);
#pragma unroll
    for (int i = 0; i < 8; ++i) {
      qa0[i] = t0[i]; qa0[i + 8] = t1[i];
      qa1[i] = t2[i]; qa1[i + 8] = t3[i];
    }
  }

  v8f o[4] = {v8f{}, v8f{}, v8f{}, v8f{}};  // O accum: 4 d-chunks of 16 cols
  float rmax[8], rsum[8];                   // rsum = per-LANE partial denominator
#pragma unroll
  for (int i = 0; i < 8; ++i) { rmax[i] = -INFINITY; rsum[i] = 0.0f; }

  const float* mrow = M + (size_t)(q0 + (hi ? 8 : 0)) * NKc + l15;

  // One 32-key step; consumes bkc (K tiles for kb), preloads kb+32 into bkn.
  auto step = [&](int kb, v16bf (&bkc)[4], v16bf (&bkn)[4]) {
    // issue next tile's K loads first (overlaps with this tile's compute;
    // the one-past-the-end preload lands harmlessly inside VT and is unused)
    load_ktile(Kb, kb + 32, l15, hi, bkn);

    // ---- S = Q*K^T + m for two 16-key halves ----
    v8f s[2];
#pragma unroll
    for (int h = 0; h < 2; ++h) {
      v8f acc = v8f{};
      acc = __builtin_amdgcn_wmma_f32_16x16x32_bf16(false, qa0, false, bkc[2 * h],
                                                    (short)0, acc, false, false);
      acc = __builtin_amdgcn_wmma_f32_16x16x32_bf16(false, qa1, false, bkc[2 * h + 1],
                                                    (short)0, acc, false, false);
      // additive mask, streamed non-temporally (read-once, keep L2 for K/V)
      const float* mp = mrow + kb + h * 16;
#pragma unroll
      for (int i = 0; i < 8; ++i)
        acc[i] += __builtin_nontemporal_load(mp + (size_t)i * NKc);
      s[h] = acc;
    }
    // prefetch ALL 16 m-rows of the tile 2 iterations ahead (lane halves cover 8 rows each)
    if (kb + 64 < NKc) {
#pragma unroll
      for (int i = 0; i < 8; ++i)
        __builtin_prefetch(mrow + (size_t)i * NKc + kb + 64, 0, 3);
    }

    // ---- online softmax: row max over 32 keys (16-lane shuffle reduce) ----
    float alpha[8];
#pragma unroll
    for (int i = 0; i < 8; ++i) {
      float t = fmaxf(s[0][i], s[1][i]);
      t = fmaxf(t, __shfl_xor(t, 1));
      t = fmaxf(t, __shfl_xor(t, 2));
      t = fmaxf(t, __shfl_xor(t, 4));
      t = fmaxf(t, __shfl_xor(t, 8));
      float mn = fmaxf(rmax[i], t);
      alpha[i] = __expf(rmax[i] - mn);
      rmax[i] = mn;
    }

    // ---- P = exp(S - mnew); stash bf16 P into LDS (C layout -> row-major) ----
    // DS ops execute in-order per wave: RAW through LDS is safe; the compiler
    // inserts s_wait_dscnt for the loaded values.
#pragma unroll
    for (int h = 0; h < 2; ++h)
#pragma unroll
      for (int i = 0; i < 8; ++i) {
        float p = __expf(s[h][i] - rmax[i]);
        s[h][i] = p;
        lds[wave][i + (hi ? 8 : 0)][h * 16 + l15] = (bf16)p;
      }

    // ---- per-lane partial denominator + rescale running state (no shuffles) ----
#pragma unroll
    for (int i = 0; i < 8; ++i) {
      rsum[i] = rsum[i] * alpha[i] + (s[0][i] + s[1][i]);
#pragma unroll
      for (int c = 0; c < 4; ++c) o[c][i] *= alpha[i];
    }

    // ---- reload P in A layout (16x32, K = keys) from LDS ----
    v16bf pa;
#pragma unroll
    for (int j = 0; j < 8; ++j) {
      int koff = 2 * j + (hi ? 16 : 0);
      pa[2 * j]     = lds[wave][l15][koff];
      pa[2 * j + 1] = lds[wave][l15][koff + 1];
    }

    // ---- O += P * V : 4 WMMAs over d-chunks, B from transposed V ----
#pragma unroll
    for (int c = 0; c < 4; ++c) {
      const bf16* vp = VT + (size_t)(c * 16 + l15) * NKc + kb + (hi ? 16 : 0);
      v16bf bv = *(const v16bf*)(vp);
      o[c] = __builtin_amdgcn_wmma_f32_16x16x32_bf16(false, pa, false, bv,
                                                     (short)0, o[c], false, false);
    }
  };

  // ping/pong K-tile register buffers; unroll-by-2 makes rotation free
  v16bf bkA[4], bkB[4];
  load_ktile(Kb, 0, l15, hi, bkA);
  for (int kb = 0; kb < NKc; kb += 64) {
    step(kb, bkA, bkB);
    step(kb + 32, bkB, bkA);
  }

  // ---- final denominator: one 16-lane reduction, then v_rcp + scale + store ----
#pragma unroll
  for (int i = 0; i < 8; ++i) {
    float r = rsum[i];
    r += __shfl_xor(r, 1);
    r += __shfl_xor(r, 2);
    r += __shfl_xor(r, 4);
    r += __shfl_xor(r, 8);
    float inv = __builtin_amdgcn_rcpf(r);
    int row = i + (hi ? 8 : 0);
#pragma unroll
    for (int c = 0; c < 4; ++c)
      out[(size_t)(q0 + row) * DD + c * 16 + l15] = o[c][i] * inv;
  }
}

// ---------------- host launcher ----------------

extern "C" void kernel_launch(void* const* d_in, const int* in_sizes, int n_in,
                              void* d_out, int out_size, void* d_ws, size_t ws_size,
                              hipStream_t stream) {
  const float* K = (const float*)d_in[0];
  const float* V = (const float*)d_in[1];
  const float* Q = (const float*)d_in[2];
  const float* M = (const float*)d_in[3];
  float* out = (float*)d_out;

  bf16* Qb = (bf16*)d_ws;                 // 1 MB
  bf16* Kb = Qb + (size_t)NQc * DD;       // 1 MB
  bf16* VT = Kb + (size_t)NKc * DD;       // 1 MB (transposed [D][NK])

  const int n = NQc * DD;                 // 524288 elements
  cvt_q <<<n / 256, 256, 0, stream>>>(Q, Qb);
  cvt_k <<<n / 256, 256, 0, stream>>>(K, Kb);
  cvt_vt<<<n / 256, 256, 0, stream>>>(V, VT);

  // 512 query tiles of 16 rows; 8 waves (tiles) per block -> 64 blocks x 256 threads
  fattn<<<NQc / 128, 256, 0, stream>>>(Qb, Kb, VT, M, out);
}
